// GDN_31911607009298
// MI455X (gfx1250) — compile-verified, matrix operand order checked by reading, not run on hardware
//
#include <hip/hip_runtime.h>
#include <hip/hip_bf16.h>
#include <cstdint>

// Problem constants (fixed by the reference)
#define N_NODES 512
#define HID     256
#define TOPK    16
#define HEADS1  8
#define BATCH   64
#define BN      (BATCH * N_NODES)   // 32768

typedef __attribute__((ext_vector_type(16))) _Float16 v16h;
typedef __attribute__((ext_vector_type(8)))  _Float16 v8h;
typedef __attribute__((ext_vector_type(8)))  float    v8f;

union HFrag { v16h v; v8h h[2]; };

// ---------------------------------------------------------------------------
// f32 -> f16 converters
// ---------------------------------------------------------------------------
__global__ void cvt_f32_f16(const float* __restrict__ src, _Float16* __restrict__ dst, int n) {
    int i = blockIdx.x * 256 + threadIdx.x;
    if (i < n) dst[i] = (_Float16)src[i];
}

// out[r*Kout + k] = W[r*K2 + off + k]   (extract column-slice of a weight, to f16)
__global__ void cvt_slice_f16(const float* __restrict__ W, _Float16* __restrict__ out,
                              int K2, int off, int Kout, int total) {
    int i = blockIdx.x * 256 + threadIdx.x;
    if (i >= total) return;
    int r = i / Kout, k = i % Kout;
    out[i] = (_Float16)W[(size_t)r * K2 + off + k];
}

// ---------------------------------------------------------------------------
// Row norms of emb_w
// ---------------------------------------------------------------------------
__global__ void row_norms(const float* __restrict__ emb, float* __restrict__ norms) {
    int n = blockIdx.x * 256 + threadIdx.x;
    if (n >= N_NODES) return;
    float s = 0.f;
    const float* e = emb + (size_t)n * HID;
    for (int k = 0; k < HID; ++k) s += e[k] * e[k];
    norms[n] = fmaxf(sqrtf(s), 1e-12f);
}

// ---------------------------------------------------------------------------
// Cosine-similarity top-k per row (one block per row)
// ---------------------------------------------------------------------------
__global__ __launch_bounds__(256) void topk_kernel(const float* __restrict__ emb,
                                                   const float* __restrict__ norms,
                                                   int* __restrict__ knn) {
    int i = blockIdx.x, t = threadIdx.x;
    __shared__ float rowi[HID];
    __shared__ float sim[N_NODES];
    __shared__ float rv[256];
    __shared__ int   ri[256];
    rowi[t] = emb[(size_t)i * HID + t];
    __syncthreads();
    float inv_i = 1.f / norms[i];
    for (int j = t; j < N_NODES; j += 256) {
        const float* ej = emb + (size_t)j * HID;
        float s = 0.f;
        for (int k = 0; k < HID; ++k) s += rowi[k] * ej[k];
        sim[j] = (j == i) ? -1e9f : s * inv_i / norms[j];
    }
    __syncthreads();
    for (int r = 0; r < TOPK; ++r) {
        float v0 = sim[t], v1 = sim[t + 256];
        float v; int id;
        if (v1 > v0) { v = v1; id = t + 256; } else { v = v0; id = t; }
        rv[t] = v; ri[t] = id;
        __syncthreads();
        for (int s = 128; s > 0; s >>= 1) {
            if (t < s && rv[t + s] > rv[t]) { rv[t] = rv[t + s]; ri[t] = ri[t + s]; }
            __syncthreads();
        }
        if (t == 0) { knn[i * TOPK + r] = ri[0]; sim[ri[0]] = -1e9f; }
        __syncthreads();
    }
}

// ---------------------------------------------------------------------------
// Fold a^T into W: wq_eff[h,k] = sum_d a1[h,d]*Wq1[h*32+d,k], etc.
// ---------------------------------------------------------------------------
__global__ void eff_weights(const float* __restrict__ Wq1, const float* __restrict__ Wk1,
                            const float* __restrict__ a1,
                            const float* __restrict__ Wq2, const float* __restrict__ Wk2,
                            const float* __restrict__ a2,
                            float* __restrict__ wq1e, float* __restrict__ wk1e,
                            float* __restrict__ wq2e, float* __restrict__ wk2e) {
    int idx = blockIdx.x * 256 + threadIdx.x;
    const int DH = HID / HEADS1;  // 32
    if (idx < HEADS1 * HID) {
        int h = idx >> 8, k = idx & 255;
        float s = 0.f;
        for (int d = 0; d < DH; ++d) s += a1[h * 2 * DH + d] * Wq1[(size_t)(h * DH + d) * HID + k];
        wq1e[idx] = s;
    } else if (idx < 2 * HEADS1 * HID) {
        int j = idx - HEADS1 * HID;
        int h = j >> 8, k = j & 255;
        float s = 0.f;
        for (int d = 0; d < DH; ++d) s += a1[h * 2 * DH + DH + d] * Wk1[(size_t)(h * DH + d) * HID + k];
        wk1e[j] = s;
    } else if (idx < 2 * HEADS1 * HID + HID) {
        int k = idx - 2 * HEADS1 * HID;
        float s = 0.f;
        for (int d = 0; d < HID; ++d) s += a2[d] * Wq2[(size_t)d * HID + k];
        wq2e[k] = s;
    } else if (idx < 2 * HEADS1 * HID + 2 * HID) {
        int k = idx - (2 * HEADS1 * HID + HID);
        float s = 0.f;
        for (int d = 0; d < HID; ++d) s += a2[HID + d] * Wk2[(size_t)d * HID + k];
        wk2e[k] = s;
    }
}

// q_s / k_s per node (batch-invariant): one block of 32 threads per node
__global__ void node_scores(const float* __restrict__ emb,
                            const float* __restrict__ wq1e, const float* __restrict__ wk1e,
                            const float* __restrict__ wq2e, const float* __restrict__ wk2e,
                            float* __restrict__ qs1, float* __restrict__ ks1,
                            float* __restrict__ qs2, float* __restrict__ ks2) {
    int n = blockIdx.x, t = threadIdx.x;
    __shared__ float e[HID];
    for (int k = t; k < HID; k += 32) e[k] = emb[(size_t)n * HID + k];
    __syncthreads();
    if (t < 8) {
        float s = 0.f; for (int k = 0; k < HID; ++k) s += e[k] * wq1e[t * HID + k];
        qs1[n * HEADS1 + t] = s;
    } else if (t < 16) {
        int h = t - 8;
        float s = 0.f; for (int k = 0; k < HID; ++k) s += e[k] * wk1e[h * HID + k];
        ks1[n * HEADS1 + h] = s;
    } else if (t == 16) {
        float s = 0.f; for (int k = 0; k < HID; ++k) s += e[k] * wq2e[k];
        qs2[n] = s;
    } else if (t == 17) {
        float s = 0.f; for (int k = 0; k < HID; ++k) s += e[k] * wk2e[k];
        ks2[n] = s;
    }
}

// Per-node softmax over 16 neighbors (batch-invariant alpha)
__global__ void alpha_kernel(const int* __restrict__ knn,
                             const float* __restrict__ qs, const float* __restrict__ ks,
                             int heads, float* __restrict__ alpha) {
    int n = blockIdx.x, t = threadIdx.x;   // blockDim = 16*heads
    int j = t % TOPK, h = t / TOPK;
    __shared__ float e[TOPK * HEADS1];
    __shared__ float mx[HEADS1];
    __shared__ float sm[HEADS1];
    int src = knn[n * TOPK + j];
    float v = qs[n * heads + h] + ks[src * heads + h];
    v = v > 0.f ? v : 0.2f * v;            // leaky_relu(0.2)
    e[t] = v;
    __syncthreads();
    if (j == 0) {
        float m = -1e30f;
        for (int q = 0; q < TOPK; ++q) m = fmaxf(m, e[h * TOPK + q]);
        float s = 0.f;
        for (int q = 0; q < TOPK; ++q) s += expf(e[h * TOPK + q] - m);
        mx[h] = m; sm[h] = s;
    }
    __syncthreads();
    alpha[(n * TOPK + j) * heads + h] = expf(v - mx[h]) / (sm[h] + 1e-8f);
}

// ---------------------------------------------------------------------------
// WMMA f16 GEMM:  C[M,256] = A[M,K] * B[256,K]^T  (+addRow[row%512]) (+bias) (relu?)
// Each wave computes a 16x64 strip: 1 A fragment reused across 4 WMMAs / k-step.
// Block = (32,8) -> 128 rows x 64 cols per block.
// ---------------------------------------------------------------------------
__global__ __launch_bounds__(256) void wmma_gemm(
    const _Float16* __restrict__ A,   // M x K row-major
    const _Float16* __restrict__ B,   // 256 x K row-major (weight rows = output cols)
    float* __restrict__ Cf,           // f32 out (or null)
    _Float16* __restrict__ Ch,        // f16 out (or null)
    const float* __restrict__ addRow, // 512 x 256 per-node additive table (or null)
    const float* __restrict__ bias,   // 256 (or null)
    int M, int K, int relu) {
    const int NCOL = HID;             // output width 256
    int lane  = threadIdx.x;          // 0..31
    int wave  = threadIdx.y;          // 0..7
    int tileN = blockIdx.x * 64;      // 64-wide N strip per wave
    int tileM = (blockIdx.y * 8 + wave) * 16;
    if (tileM >= M) return;

    int m  = lane & 15;
    int hs = lane >> 4;               // which K half-group this lane holds
    const _Float16* aRow  = A + (size_t)(tileM + m) * K + hs * 8;
    const _Float16* bRow0 = B + (size_t)(tileN + m) * K + hs * 8;

    v8f c0 = {}, c1 = {}, c2 = {}, c3 = {};
    for (int k = 0; k < K; k += 32) {
        HFrag a, b0, b1, b2, b3;
        a.h[0]  = *(const v8h*)(aRow + k);
        a.h[1]  = *(const v8h*)(aRow + k + 16);
        b0.h[0] = *(const v8h*)(bRow0 + k);
        b0.h[1] = *(const v8h*)(bRow0 + k + 16);
        b1.h[0] = *(const v8h*)(bRow0 + 16 * K + k);
        b1.h[1] = *(const v8h*)(bRow0 + 16 * K + k + 16);
        b2.h[0] = *(const v8h*)(bRow0 + 32 * K + k);
        b2.h[1] = *(const v8h*)(bRow0 + 32 * K + k + 16);
        b3.h[0] = *(const v8h*)(bRow0 + 48 * K + k);
        b3.h[1] = *(const v8h*)(bRow0 + 48 * K + k + 16);
        c0 = __builtin_amdgcn_wmma_f32_16x16x32_f16(false, a.v, false, b0.v, (short)0, c0, false, false);
        c1 = __builtin_amdgcn_wmma_f32_16x16x32_f16(false, a.v, false, b1.v, (short)0, c1, false, false);
        c2 = __builtin_amdgcn_wmma_f32_16x16x32_f16(false, a.v, false, b2.v, (short)0, c2, false, false);
        c3 = __builtin_amdgcn_wmma_f32_16x16x32_f16(false, a.v, false, b3.v, (short)0, c3, false, false);
    }

    int row0 = tileM + hs * 8;
    v8f acc[4] = {c0, c1, c2, c3};
    for (int t = 0; t < 4; ++t) {
        int col = tileN + t * 16 + m;
        for (int r = 0; r < 8; ++r) {
            int row = row0 + r;
            float v = acc[t][r];
            if (addRow) v += addRow[(size_t)(row & (N_NODES - 1)) * NCOL + col];
            if (bias)   v += bias[col];
            if (relu)   v = v > 0.f ? v : 0.f;
            if (Cf) Cf[(size_t)row * NCOL + col] = v;
            if (Ch) Ch[(size_t)row * NCOL + col] = (_Float16)v;
        }
    }
}

// ---------------------------------------------------------------------------
// Graph aggregation + elu + LayerNorm (one block = one output row, 256 threads)
// ---------------------------------------------------------------------------
__global__ __launch_bounds__(256) void agg_elu_ln(
    const float* __restrict__ V,      // BN x 256 (f32)
    const int* __restrict__ knn,      // 512 x 16
    const float* __restrict__ alpha,  // (512*16) x heads
    int heads,
    const float* __restrict__ g, const float* __restrict__ b,
    _Float16* __restrict__ outH) {    // BN x 256 (f16 for next GEMM)
    int i  = blockIdx.x;
    int n  = i & (N_NODES - 1);
    int bb = i >> 9;
    int t  = threadIdx.x;
    __shared__ int   s_src[TOPK];
    __shared__ float s_alpha[TOPK * HEADS1];
    __shared__ float red[256];
    if (t < TOPK) s_src[t] = knn[n * TOPK + t];
    if (t < TOPK * heads) s_alpha[t] = alpha[(size_t)n * TOPK * heads + t];
    __syncthreads();
    // prefetch the 16 gathered rows (global_prefetch_b8): thread t covers
    // neighbour j = t>>3, 128B line chunk = t&7  (16 rows x 1KB)
    {
        int j = t >> 3, chunk = t & 7;
        if (t < TOPK * 8)
            __builtin_prefetch(&V[((size_t)(bb * N_NODES + s_src[j])) * HID + chunk * 32], 0, 0);
    }
    int h = (t * heads) >> 8;         // heads=8 -> t/32 ; heads=1 -> 0
    float acc = 0.f;
    for (int j = 0; j < TOPK; ++j) {
        int src = s_src[j];
        acc += s_alpha[j * heads + h] * V[((size_t)(bb * N_NODES + src)) * HID + t];
    }
    float z = acc > 0.f ? acc : (expf(acc) - 1.f);   // elu
    red[t] = z; __syncthreads();
    for (int s = 128; s > 0; s >>= 1) { if (t < s) red[t] += red[t + s]; __syncthreads(); }
    float mu = red[0] * (1.f / HID);
    __syncthreads();
    float d = z - mu;
    red[t] = d * d; __syncthreads();
    for (int s = 128; s > 0; s >>= 1) { if (t < s) red[t] += red[t + s]; __syncthreads(); }
    float var = red[0] * (1.f / HID);
    float o = d * rsqrtf(var + 1e-5f) * g[t] + b[t];
    outH[(size_t)i * HID + t] = (_Float16)o;
}

// ---------------------------------------------------------------------------
// Final projection: out[row] = dot(h[row], Wf2) + bf2   (wave per row)
// ---------------------------------------------------------------------------
__global__ __launch_bounds__(256) void final_dot(const float* __restrict__ H,
                                                 const float* __restrict__ Wf2,
                                                 const float* __restrict__ bf2,
                                                 float* __restrict__ out, int M) {
    int row = blockIdx.x * 8 + threadIdx.y;
    if (row >= M) return;
    int lane = threadIdx.x;
    const float* h = H + (size_t)row * HID;
    float s = 0.f;
    for (int k = lane; k < HID; k += 32) s += h[k] * Wf2[k];
    for (int off = 16; off > 0; off >>= 1) s += __shfl_down(s, off, 32);
    if (lane == 0) out[row] = s + bf2[0];
}

// ---------------------------------------------------------------------------
extern "C" void kernel_launch(void* const* d_in, const int* in_sizes, int n_in,
                              void* d_out, int out_size, void* d_ws, size_t ws_size,
                              hipStream_t stream) {
    const float* x    = (const float*)d_in[0];
    const float* embw = (const float*)d_in[1];
    const float* Wq1  = (const float*)d_in[2];
    const float* Wk1  = (const float*)d_in[3];
    const float* Wv1  = (const float*)d_in[4];
    const float* a1   = (const float*)d_in[5];
    const float* g1   = (const float*)d_in[6];
    const float* b1   = (const float*)d_in[7];
    const float* Wq2  = (const float*)d_in[8];
    const float* Wk2  = (const float*)d_in[9];
    const float* Wv2  = (const float*)d_in[10];
    const float* a2   = (const float*)d_in[11];
    const float* g2   = (const float*)d_in[12];
    const float* b2   = (const float*)d_in[13];
    const float* Wf1  = (const float*)d_in[14];
    const float* bf1  = (const float*)d_in[15];
    const float* Wf2  = (const float*)d_in[16];
    const float* bf2  = (const float*)d_in[17];
    float* out = (float*)d_out;

    // ---- workspace bump allocator ----
    char* ws = (char*)d_ws;
    size_t off = 0;
    auto alloc = [&](size_t bytes) -> char* {
        char* p = ws + off;
        off = (off + bytes + 255) & ~(size_t)255;
        return p;
    };
    int*       knn    = (int*)      alloc(N_NODES * TOPK * 4);
    float*     norms  = (float*)    alloc(N_NODES * 4);
    float*     wq1e   = (float*)    alloc(HEADS1 * HID * 4);
    float*     wk1e   = (float*)    alloc(HEADS1 * HID * 4);
    float*     wq2e   = (float*)    alloc(HID * 4);
    float*     wk2e   = (float*)    alloc(HID * 4);
    float*     qs1    = (float*)    alloc(N_NODES * HEADS1 * 4);
    float*     ks1    = (float*)    alloc(N_NODES * HEADS1 * 4);
    float*     qs2    = (float*)    alloc(N_NODES * 4);
    float*     ks2    = (float*)    alloc(N_NODES * 4);
    float*     alpha1 = (float*)    alloc(N_NODES * TOPK * HEADS1 * 4);
    float*     alpha2 = (float*)    alloc(N_NODES * TOPK * 4);
    _Float16*  emb_h  = (_Float16*) alloc(N_NODES * HID * 2);
    _Float16*  Wv1a_h = (_Float16*) alloc(HID * HID * 2);
    _Float16*  Wv1b_h = (_Float16*) alloc(HID * HID * 2);
    _Float16*  Wv2a_h = (_Float16*) alloc(HID * HID * 2);
    _Float16*  Wv2b_h = (_Float16*) alloc(HID * HID * 2);
    _Float16*  Wf1a_h = (_Float16*) alloc(HID * HID * 2);
    _Float16*  Wf1b_h = (_Float16*) alloc(HID * HID * 2);
    float*     Vc1    = (float*)    alloc(N_NODES * HID * 4);
    float*     Vc2    = (float*)    alloc(N_NODES * HID * 4);
    float*     He     = (float*)    alloc(N_NODES * HID * 4);
    _Float16*  x_h    = (_Float16*) alloc((size_t)BN * HID * 2);
    _Float16*  zh     = (_Float16*) alloc((size_t)BN * HID * 2);  // z1 then z2 (f16)
    float*     BIG    = (float*)    alloc((size_t)BN * HID * 4);  // V1 -> V2 -> h

    dim3 blk256(256);
    // --- conversions ---
    cvt_f32_f16<<<(BN * HID + 255) / 256, blk256, 0, stream>>>(x, x_h, BN * HID);
    cvt_f32_f16<<<(N_NODES * HID + 255) / 256, blk256, 0, stream>>>(embw, emb_h, N_NODES * HID);
    int totW = HID * HID;
    cvt_slice_f16<<<(totW + 255) / 256, blk256, 0, stream>>>(Wv1, Wv1a_h, 2 * HID, 0,   HID, totW);
    cvt_slice_f16<<<(totW + 255) / 256, blk256, 0, stream>>>(Wv1, Wv1b_h, 2 * HID, HID, HID, totW);
    cvt_slice_f16<<<(totW + 255) / 256, blk256, 0, stream>>>(Wv2, Wv2a_h, 2 * HID, 0,   HID, totW);
    cvt_slice_f16<<<(totW + 255) / 256, blk256, 0, stream>>>(Wv2, Wv2b_h, 2 * HID, HID, HID, totW);
    cvt_slice_f16<<<(totW + 255) / 256, blk256, 0, stream>>>(Wf1, Wf1a_h, 2 * HID, 0,   HID, totW);
    cvt_slice_f16<<<(totW + 255) / 256, blk256, 0, stream>>>(Wf1, Wf1b_h, 2 * HID, HID, HID, totW);

    // --- batch-invariant graph/attention precompute ---
    row_norms<<<2, blk256, 0, stream>>>(embw, norms);
    topk_kernel<<<N_NODES, blk256, 0, stream>>>(embw, norms, knn);
    eff_weights<<<(2 * HEADS1 * HID + 2 * HID + 255) / 256, blk256, 0, stream>>>(
        Wq1, Wk1, a1, Wq2, Wk2, a2, wq1e, wk1e, wq2e, wk2e);
    node_scores<<<N_NODES, dim3(32), 0, stream>>>(embw, wq1e, wk1e, wq2e, wk2e, qs1, ks1, qs2, ks2);
    alpha_kernel<<<N_NODES, dim3(TOPK * HEADS1), 0, stream>>>(knn, qs1, ks1, HEADS1, alpha1);
    alpha_kernel<<<N_NODES, dim3(TOPK), 0, stream>>>(knn, qs2, ks2, 1, alpha2);

    // batch-invariant emb-halves of the concat GEMMs (WMMA, M=512)
    dim3 gblk(32, 8);
    dim3 gridC(HID / 64, N_NODES / 128);
    wmma_gemm<<<gridC, gblk, 0, stream>>>(emb_h, Wv1b_h, Vc1, nullptr, nullptr, nullptr, N_NODES, HID, 0);
    wmma_gemm<<<gridC, gblk, 0, stream>>>(emb_h, Wv2b_h, Vc2, nullptr, nullptr, nullptr, N_NODES, HID, 0);
    wmma_gemm<<<gridC, gblk, 0, stream>>>(emb_h, Wf1b_h, He,  nullptr, nullptr, nullptr, N_NODES, HID, 0);

    // --- per-batch heavy pipeline ---
    dim3 gridB(HID / 64, BN / 128);
    // V1 = x @ Wv1[:, :256]^T + Vc1[node]
    wmma_gemm<<<gridB, gblk, 0, stream>>>(x_h, Wv1a_h, BIG, nullptr, Vc1, nullptr, BN, HID, 0);
    // z1 = LN(elu(segment_sum(alpha1 * V1[src])))  (f16)
    agg_elu_ln<<<BN, blk256, 0, stream>>>(BIG, knn, alpha1, HEADS1, g1, b1, zh);
    // V2 = z1 @ Wv2[:, :256]^T + Vc2[node]
    wmma_gemm<<<gridB, gblk, 0, stream>>>(zh, Wv2a_h, BIG, nullptr, Vc2, nullptr, BN, HID, 0);
    // z2 = LN(elu(segment_sum(alpha2 * V2[src])))  (f16)
    agg_elu_ln<<<BN, blk256, 0, stream>>>(BIG, knn, alpha2, 1, g2, b2, zh);
    // h = relu(z2 @ Wf1[:, :256]^T + He[node] + bf1)  (f32)
    wmma_gemm<<<gridB, gblk, 0, stream>>>(zh, Wf1a_h, BIG, nullptr, He, bf1, BN, HID, 1);
    // out = h @ Wf2^T + bf2
    final_dot<<<BN / 8, dim3(32, 8), 0, stream>>>(BIG, Wf2, bf2, out, BN);

    (void)in_sizes; (void)n_in; (void)out_size; (void)ws_size;
}
